// LeNet5_2AFRNN_CIFAR10_52802327937305
// MI455X (gfx1250) — matrix operand order repaired and checked by reading, not compiled, not run on hardware
//
#include <hip/hip_runtime.h>
#include <hip/hip_bf16.h>
#include <math.h>

typedef __attribute__((ext_vector_type(16))) _Float16 v16h;
typedef __attribute__((ext_vector_type(8)))  float    v8f;

// ---------------------------------------------------------------------------
// Fuzzy conv layers: Lukasiewicz t-norm (Yager p=1):
//   out = max(0, max_L(x_l + w_l) - 1), fused with 2x2 maxpool where present.
// ---------------------------------------------------------------------------

// conv1 + pool: x[128,3,32,32], w1[6,3,5,5] -> p1[128,6,14,14]
__global__ void conv1pool_kernel(const float* __restrict__ x,
                                 const float* __restrict__ w1,
                                 float* __restrict__ p1) {
    const int blk = blockIdx.x;          // b*6 + o
    const int b = blk / 6, o = blk % 6;
    __shared__ float sw[75];
    for (int i = threadIdx.x; i < 75; i += blockDim.x) sw[i] = w1[o * 75 + i];
    __syncthreads();
    const int t = threadIdx.x;
    if (t >= 196) return;
    const int oy = t / 14, ox = t % 14;
    const float* xb = x + (size_t)b * 3 * 32 * 32;
    float m = -1e30f;
    #pragma unroll
    for (int py = 0; py < 2; ++py)
    #pragma unroll
    for (int px = 0; px < 2; ++px) {
        const int iy0 = oy * 2 + py, ix0 = ox * 2 + px;   // conv out pos [0,28)
        for (int c = 0; c < 3; ++c)
            #pragma unroll
            for (int kh = 0; kh < 5; ++kh)
                #pragma unroll
                for (int kw = 0; kw < 5; ++kw) {
                    float xv = xb[(c * 32 + iy0 + kh) * 32 + ix0 + kw];
                    float wv = sw[(c * 5 + kh) * 5 + kw];
                    m = fmaxf(m, xv + wv);
                }
    }
    p1[(((size_t)b * 6 + o) * 14 + oy) * 14 + ox] = fmaxf(0.f, m - 1.f);
}

// conv2 + pool: p1[128,6,14,14], w2[16,6,5,5] -> p2[128,16,5,5]
__global__ void conv2pool_kernel(const float* __restrict__ p1,
                                 const float* __restrict__ w2,
                                 float* __restrict__ p2) {
    const int b = blockIdx.x;
    __shared__ float sx[6 * 14 * 14];    // 1176 floats
    __shared__ float sw[16 * 150];       // 2400 floats
    for (int i = threadIdx.x; i < 1176; i += blockDim.x) sx[i] = p1[(size_t)b * 1176 + i];
    for (int i = threadIdx.x; i < 2400; i += blockDim.x) sw[i] = w2[i];
    __syncthreads();
    const int t = threadIdx.x;
    if (t >= 400) return;
    const int o = t / 25, pos = t % 25;
    const int oy = pos / 5, ox = pos % 5;
    const float* wo = sw + o * 150;
    float m = -1e30f;
    #pragma unroll
    for (int py = 0; py < 2; ++py)
    #pragma unroll
    for (int px = 0; px < 2; ++px) {
        const int iy0 = oy * 2 + py, ix0 = ox * 2 + px;   // conv out pos [0,10)
        for (int c = 0; c < 6; ++c)
            #pragma unroll
            for (int kh = 0; kh < 5; ++kh)
                #pragma unroll
                for (int kw = 0; kw < 5; ++kw)
                    m = fmaxf(m, sx[(c * 14 + iy0 + kh) * 14 + ix0 + kw] +
                                 wo[(c * 5 + kh) * 5 + kw]);
    }
    p2[((size_t)b * 16 + o) * 25 + pos] = fmaxf(0.f, m - 1.f);
}

// conv3 (1x1 spatial): p2[128,16,5,5], w3[120,16,5,5] -> y3[128,120]
__global__ void conv3_kernel(const float* __restrict__ p2,
                             const float* __restrict__ w3,
                             float* __restrict__ y3) {
    const int b = blockIdx.x;
    __shared__ float sx[400];
    for (int i = threadIdx.x; i < 400; i += blockDim.x) sx[i] = p2[(size_t)b * 400 + i];
    __syncthreads();
    const int o = threadIdx.x;
    if (o >= 120) return;
    const float* wo = w3 + (size_t)o * 400;
    float m = -1e30f;
    for (int l = 0; l < 400; ++l) m = fmaxf(m, sx[l] + wo[l]);
    y3[(size_t)b * 120 + o] = fmaxf(0.f, m - 1.f);
}

// ---------------------------------------------------------------------------
// f16 packing for WMMA
// A side stays row-major (per-lane K runs are contiguous -> b128 loads).
// B side is pre-swizzled into WMMA B-fragment order so each lane loads one
// contiguous 32-byte chunk:   Bf[((tileN*nkt + kt)*32 + lane)*16 + e]
// with element e (0..15): kk = kt*32 + 16*(lane>>4) + 2*(e>>1) + (e&1),
//                         col = tileN*16 + (lane&15).
// ---------------------------------------------------------------------------

// Ah[128][128] <- y3[128][120], pad K (row-major)
__global__ void cvtA_kernel(const float* __restrict__ y3, _Float16* __restrict__ Ah) {
    int i = blockIdx.x * blockDim.x + threadIdx.x;
    if (i >= 128 * 128) return;
    int b = i >> 7, k = i & 127;
    Ah[i] = (k < 120) ? (_Float16)y3[b * 120 + k] : (_Float16)0.f;
}

// Wdh: packed B fragments for GEMM1. wd[120][84] -> 6 tileN x 4 kt x 32 lanes x 16
__global__ void cvtWd_kernel(const float* __restrict__ wd, _Float16* __restrict__ Bf) {
    int i = blockIdx.x * blockDim.x + threadIdx.x;
    if (i >= 6 * 4 * 32 * 16) return;          // 12288
    int e     = i & 15;
    int lane  = (i >> 4) & 31;
    int kt    = (i >> 9) & 3;
    int tileN = i >> 11;
    int half = lane >> 4, l16 = lane & 15;
    int kk = kt * 32 + 16 * half + 2 * (e >> 1) + (e & 1);
    int n  = tileN * 16 + l16;
    Bf[i] = (kk < 120 && n < 84) ? (_Float16)wd[kk * 84 + n] : (_Float16)0.f;
}

// Wfh: packed B fragments for GEMM2. wf[84][10] -> 1 tileN x 3 kt x 32 lanes x 16
__global__ void cvtWf_kernel(const float* __restrict__ wf, _Float16* __restrict__ Bf) {
    int i = blockIdx.x * blockDim.x + threadIdx.x;
    if (i >= 3 * 32 * 16) return;              // 1536
    int e    = i & 15;
    int lane = (i >> 4) & 31;
    int kt   = i >> 9;
    int half = lane >> 4, l16 = lane & 15;
    int kk = kt * 32 + 16 * half + 2 * (e >> 1) + (e & 1);
    Bf[i] = (kk < 84 && l16 < 10) ? (_Float16)wf[kk * 10 + l16] : (_Float16)0.f;
}

// ---------------------------------------------------------------------------
// A-fragment loader, contiguous K order matching the packed B (wave32):
//   A 16x32 f16: lane half h, vgpr j: K = 16h + 2*j
//   (one contiguous 16-half run per lane -> compiler emits 2x b128)
// ---------------------------------------------------------------------------
__device__ inline v16h load_a_frag(const _Float16* __restrict__ A, int lda,
                                   int row, int kb, int half) {
    v16h a;
    #pragma unroll
    for (int j = 0; j < 8; ++j) {
        int kk = kb + 16 * half + 2 * j;
        a[2 * j]     = A[row * lda + kk];
        a[2 * j + 1] = A[row * lda + kk + 1];
    }
    return a;
}
__device__ inline v16h load_b_frag_packed(const _Float16* __restrict__ Bf,
                                          int frag_idx, int lane) {
    return *(const v16h*)(Bf + ((size_t)frag_idx * 32 + lane) * 16);
}

// GEMM1: Zh[128][96] = tanh(Ah[128][128] x Wd) ; 1 wave per 16x16 tile
__global__ void gemm1_wmma_kernel(const _Float16* __restrict__ Ah,
                                  const _Float16* __restrict__ Wdf,
                                  _Float16* __restrict__ Zh) {
    const int tileM = blockIdx.x / 6, tileN = blockIdx.x % 6;
    const int lane = threadIdx.x;
    const int half = lane >> 4, l16 = lane & 15;
    v8f c = {};
    #pragma unroll
    for (int kt = 0; kt < 4; ++kt) {
        v16h a = load_a_frag(Ah, 128, tileM * 16 + l16, kt * 32, half);
        v16h b = load_b_frag_packed(Wdf, tileN * 4 + kt, lane);
        c = __builtin_amdgcn_wmma_f32_16x16x32_f16(false, a, false, b,
                                                   (short)0, c, false, false);
    }
    #pragma unroll
    for (int r = 0; r < 8; ++r) {
        int row = tileM * 16 + r + 8 * half;
        int col = tileN * 16 + l16;
        Zh[row * 96 + col] = (_Float16)tanhf(c[r]);
    }
}

// GEMM2: logits[128][16] = Zh[128][96] x Wf ; 1 wave per M-tile
__global__ void gemm2_wmma_kernel(const _Float16* __restrict__ Zh,
                                  const _Float16* __restrict__ Wff,
                                  float* __restrict__ logits) {
    const int tileM = blockIdx.x;
    const int lane = threadIdx.x;
    const int half = lane >> 4, l16 = lane & 15;
    v8f c = {};
    #pragma unroll
    for (int kt = 0; kt < 3; ++kt) {
        v16h a = load_a_frag(Zh, 96, tileM * 16 + l16, kt * 32, half);
        v16h b = load_b_frag_packed(Wff, kt, lane);
        c = __builtin_amdgcn_wmma_f32_16x16x32_f16(false, a, false, b,
                                                   (short)0, c, false, false);
    }
    #pragma unroll
    for (int r = 0; r < 8; ++r) {
        int row = tileM * 16 + r + 8 * half;
        logits[row * 16 + l16] = c[r];
    }
}

// bias + log_softmax: out[128][10]
__global__ void logsoftmax_kernel(const float* __restrict__ logits,
                                  const float* __restrict__ bf,
                                  float* __restrict__ out) {
    int b = blockIdx.x * blockDim.x + threadIdx.x;
    if (b >= 128) return;
    float v[10], mx = -1e30f;
    #pragma unroll
    for (int j = 0; j < 10; ++j) {
        v[j] = logits[b * 16 + j] + bf[j];
        mx = fmaxf(mx, v[j]);
    }
    float s = 0.f;
    #pragma unroll
    for (int j = 0; j < 10; ++j) s += __expf(v[j] - mx);
    float ls = __logf(s) + mx;
    #pragma unroll
    for (int j = 0; j < 10; ++j) out[b * 10 + j] = v[j] - ls;
}

// ---------------------------------------------------------------------------
// Launch
// ---------------------------------------------------------------------------
extern "C" void kernel_launch(void* const* d_in, const int* in_sizes, int n_in,
                              void* d_out, int out_size, void* d_ws, size_t ws_size,
                              hipStream_t stream) {
    (void)in_sizes; (void)n_in; (void)out_size; (void)ws_size;
    const float* x  = (const float*)d_in[0];  // [128,3,32,32]
    const float* w1 = (const float*)d_in[1];  // [6,3,5,5]
    const float* w2 = (const float*)d_in[2];  // [16,6,5,5]
    const float* w3 = (const float*)d_in[3];  // [120,16,5,5]
    const float* wd = (const float*)d_in[4];  // [120,84]
    const float* wf = (const float*)d_in[5];  // [84,10]
    const float* bf = (const float*)d_in[6];  // [10]
    float* out = (float*)d_out;               // [128,10]

    char* wsb = (char*)d_ws;                  // byte offsets (256B aligned)
    float*    p1     = (float*)(wsb + 0);          // 128*6*14*14  = 602112 B
    float*    p2     = (float*)(wsb + 602112);     // 128*16*5*5   = 204800 B
    float*    y3     = (float*)(wsb + 806912);     // 128*120      =  61440 B
    _Float16* Ah     = (_Float16*)(wsb + 868352);  // 128*128 f16  =  32768 B
    _Float16* Wdf    = (_Float16*)(wsb + 901120);  // 6*4*32*16 f16=  24576 B
    _Float16* Wff    = (_Float16*)(wsb + 925696);  // 3*32*16  f16 =   3072 B
    _Float16* Zh     = (_Float16*)(wsb + 928768);  // 128*96   f16 =  24576 B
    float*    logits = (float*)(wsb + 953344);     // 128*16       =   8192 B

    conv1pool_kernel<<<128 * 6, 256, 0, stream>>>(x, w1, p1);
    conv2pool_kernel<<<128, 512, 0, stream>>>(p1, w2, p2);
    conv3_kernel<<<128, 128, 0, stream>>>(p2, w3, y3);

    cvtA_kernel <<<(128 * 128 + 255) / 256, 256, 0, stream>>>(y3, Ah);
    cvtWd_kernel<<<(6 * 4 * 32 * 16 + 255) / 256, 256, 0, stream>>>(wd, Wdf);
    cvtWf_kernel<<<(3 * 32 * 16 + 255) / 256, 256, 0, stream>>>(wf, Wff);

    gemm1_wmma_kernel<<<48, 32, 0, stream>>>(Ah, Wdf, Zh);   // 8 M-tiles x 6 N-tiles
    gemm2_wmma_kernel<<<8, 32, 0, stream>>>(Zh, Wff, logits);
    logsoftmax_kernel<<<1, 128, 0, stream>>>(logits, bf, out);
}
